// GINNet_35914516529230
// MI455X (gfx1250) — matrix operand order repaired and checked by reading, not compiled, not run on hardware
//
#include <hip/hip_runtime.h>
#include <hip/hip_bf16.h>

// ---------------------------------------------------------------------------
// GIN forward for MI455X (gfx1250, wave32, WMMA).
// Edge MLP (210 GFLOP) on v_wmma_f32_16x16x32_f16 with M=32 tiles and
// register double-buffered B to overlap L2 loads with the tensor pipe.
// ---------------------------------------------------------------------------

typedef _Float16 v16h __attribute__((ext_vector_type(16)));
typedef _Float16 v8h  __attribute__((ext_vector_type(8)));
typedef float    v8f  __attribute__((ext_vector_type(8)));

union V16 { v16h v; v8h h[2]; };

#define IN_DIM 128
#define HID    256
#define NCLS   40
#define NPAD   48
#define EL_HID 2048

// ---------------------------------------------------------------------------
// Pack a row-major f32 weight matrix W[K][Nreal] into f16 WMMA-B layout:
// out[((nc*(K/32)+kc)*32 + lane)*16 + i] = f16( W[kc*32 + (i&7) + 8*(lane>>4) + 16*(i>>3)][nc*16 + (lane&15)] )
// ---------------------------------------------------------------------------
__global__ __launch_bounds__(256) void pack_b_kernel(const float* __restrict__ W,
                                                     _Float16* __restrict__ out,
                                                     int K, int Nreal, int Npad) {
  int idx = blockIdx.x * 256 + threadIdx.x;
  int total = K * Npad;
  if (idx >= total) return;
  int i    = idx & 15;
  int lane = (idx >> 4) & 31;
  int blk  = idx >> 9;
  int KC   = K >> 5;
  int kc   = blk % KC;
  int nc   = blk / KC;
  int k = kc * 32 + (i & 7) + 8 * (lane >> 4) + 16 * (i >> 3);
  int n = nc * 16 + (lane & 15);
  float v = (n < Nreal) ? W[(size_t)k * Nreal + n] : 0.0f;
  out[idx] = (_Float16)v;
}

// ---------------------------------------------------------------------------
// Edge MLP: ew[e] = relu([h[row]||h[col]] @ We1 + be1) @ We2 + be2
// 128 edges/block, 4 waves; each wave owns TWO 16-edge tiles (M=32) so every
// B fragment feeds 2 WMMAs. B blocks double-buffered in registers: block nc+1
// loads issue before block nc's 16 WMMAs, hiding L2 latency behind compute.
// ---------------------------------------------------------------------------
__global__ __launch_bounds__(128) void edge_mlp_kernel(
    const float* __restrict__ h, const int* __restrict__ row, const int* __restrict__ col,
    const _Float16* __restrict__ Bpk, const float* __restrict__ be1,
    const float* __restrict__ We2, const float* __restrict__ be2,
    float* __restrict__ ew, int E) {
  __shared__ _Float16 aLds[128 * 256];   // 64 KB
  const int t  = threadIdx.x;
  const int e0 = blockIdx.x * 128;

  // ---- gather: 1 thread per edge (row half + col half), f32 -> f16 ----
  {
    int e = e0 + t;
    int nr = 0, ncn = 0;
    if (e < E) { nr = row[e]; ncn = col[e]; }
    const float4* sr = (const float4*)(h + (size_t)nr * IN_DIM);
    const float4* sc = (const float4*)(h + (size_t)ncn * IN_DIM);
    _Float16* dst = &aLds[t * 256];
#pragma unroll
    for (int j = 0; j < IN_DIM / 4; ++j) {
      float4 v = sr[j];
      dst[4 * j + 0] = (_Float16)v.x;
      dst[4 * j + 1] = (_Float16)v.y;
      dst[4 * j + 2] = (_Float16)v.z;
      dst[4 * j + 3] = (_Float16)v.w;
    }
#pragma unroll
    for (int j = 0; j < IN_DIM / 4; ++j) {
      float4 v = sc[j];
      dst[IN_DIM + 4 * j + 0] = (_Float16)v.x;
      dst[IN_DIM + 4 * j + 1] = (_Float16)v.y;
      dst[IN_DIM + 4 * j + 2] = (_Float16)v.z;
      dst[IN_DIM + 4 * j + 3] = (_Float16)v.w;
    }
  }
  __syncthreads();

  const int lane = t & 31, wave = t >> 5;
  const int g = lane >> 4, mr = lane & 15;
  const _Float16* aRow0 = &aLds[(wave * 32 + mr) * 256];
  const _Float16* aRow1 = &aLds[(wave * 32 + 16 + mr) * 256];

  // hoist all A fragments for both tiles (K=256 -> 8 chunks)
  V16 a0[8], a1[8];
#pragma unroll
  for (int kc = 0; kc < 8; ++kc) {
    a0[kc].h[0] = *(const v8h*)(aRow0 + kc * 32 + 8 * g);
    a0[kc].h[1] = *(const v8h*)(aRow0 + kc * 32 + 16 + 8 * g);
    a1[kc].h[0] = *(const v8h*)(aRow1 + kc * 32 + 8 * g);
    a1[kc].h[1] = *(const v8h*)(aRow1 + kc * 32 + 16 + 8 * g);
  }

  float acc0[8] = {0.f, 0.f, 0.f, 0.f, 0.f, 0.f, 0.f, 0.f};
  float acc1[8] = {0.f, 0.f, 0.f, 0.f, 0.f, 0.f, 0.f, 0.f};
  const _Float16* bptr = Bpk + (size_t)lane * 16;

  auto loadBlock = [&](V16* bbuf, int ncb) {
    const _Float16* nb = bptr + (size_t)ncb * 4096;   // 8 frags * 512 halves
#pragma unroll
    for (int kc = 0; kc < 8; ++kc) {
      bbuf[kc].h[0] = *(const v8h*)(nb + kc * 512);
      bbuf[kc].h[1] = *(const v8h*)(nb + kc * 512 + 8);
    }
  };
  auto computeBlock = [&](const V16* bbuf, float bias, float w2) {
    v8f c0 = {}, c1 = {};
#pragma unroll
    for (int kc = 0; kc < 8; ++kc) {
      c0 = __builtin_amdgcn_wmma_f32_16x16x32_f16(false, a0[kc].v, false, bbuf[kc].v,
                                                  (short)0, c0, false, false);
      c1 = __builtin_amdgcn_wmma_f32_16x16x32_f16(false, a1[kc].v, false, bbuf[kc].v,
                                                  (short)0, c1, false, false);
    }
#pragma unroll
    for (int r = 0; r < 8; ++r) {
      float v0 = c0[r] + bias;
      acc0[r] += (v0 > 0.f ? v0 : 0.f) * w2;
      float v1 = c1[r] + bias;
      acc1[r] += (v1 > 0.f ? v1 : 0.f) * w2;
    }
  };

  V16 bbA[8], bbB[8];
  loadBlock(bbA, 0);
  float biasA = be1[mr], w2A = We2[mr];
  for (int nc = 0; nc < EL_HID / 16; nc += 2) {
    // prefetch odd block + its scalars, then compute even block
    float biasB = be1[(nc + 1) * 16 + mr];
    float w2B   = We2[(nc + 1) * 16 + mr];
    loadBlock(bbB, nc + 1);
    computeBlock(bbA, biasA, w2A);
    // prefetch next even block, then compute odd block
    float biasN = 0.f, w2N = 0.f;
    if (nc + 2 < EL_HID / 16) {
      biasN = be1[(nc + 2) * 16 + mr];
      w2N   = We2[(nc + 2) * 16 + mr];
      loadBlock(bbA, nc + 2);
    }
    computeBlock(bbB, biasB, w2B);
    biasA = biasN; w2A = w2N;
  }

  // reduce over the 16 N-lanes within each half-wave
#pragma unroll
  for (int r = 0; r < 8; ++r) {
    float a = acc0[r], b = acc1[r];
    for (int off = 1; off < 16; off <<= 1) {
      a += __shfl_xor(a, off, 32);
      b += __shfl_xor(b, off, 32);
    }
    acc0[r] = a; acc1[r] = b;
  }
  if (mr == 0) {
    float bb2 = be2[0];
#pragma unroll
    for (int r = 0; r < 8; ++r) {
      int e = e0 + wave * 32 + r + 8 * g;
      if (e < E) ew[e] = acc0[r] + bb2;
      int e2 = e + 16;
      if (e2 < E) ew[e2] = acc1[r] + bb2;
    }
  }
}

// ---------------------------------------------------------------------------
// Deterministic single-block mean / unbiased variance of ew.
// ---------------------------------------------------------------------------
__global__ __launch_bounds__(1024) void stats_kernel(const float* __restrict__ ew,
                                                     float* __restrict__ stats, int E) {
  __shared__ float s1[1024], s2[1024];
  int t = threadIdx.x;
  float a = 0.f, b = 0.f;
  for (int i = t; i < E; i += 1024) { float v = ew[i]; a += v; b += v * v; }
  s1[t] = a; s2[t] = b;
  __syncthreads();
  for (int off = 512; off > 0; off >>= 1) {
    if (t < off) { s1[t] += s1[t + off]; s2[t] += s2[t + off]; }
    __syncthreads();
  }
  if (t == 0) {
    float mean = s1[0] / (float)E;
    float var  = (s2[0] - mean * s1[0]) / (float)(E - 1);
    stats[0] = mean;
    stats[1] = var;
  }
}

// ---------------------------------------------------------------------------
// Straight-through hard masks; wn = (ew-mean)/sqrt(var); edge_weight = wn/100 + 1.
// ---------------------------------------------------------------------------
__global__ __launch_bounds__(256) void mask_kernel(
    const float* __restrict__ ew, const float* __restrict__ stats,
    const int* __restrict__ row, const float* __restrict__ thres0,
    const float* __restrict__ thres1, float* __restrict__ em0,
    float* __restrict__ em1, int E) {
  int e = blockIdx.x * 256 + threadIdx.x;
  if (e >= E) return;
  float mean = stats[0], var = stats[1];
  float wn = (ew[e] - mean) * rsqrtf(var);
  float edge_weight = wn * 0.01f + 1.0f;
  int rr = row[e];
  float t0 = thres0[rr];
  float tt0 = 0.05f * (t0 * t0 * t0 + 5.f * t0);
  float y0 = 1.f / (1.f + __expf(-(wn - tt0) * 10.f));   // TAU = 0.1
  float m0 = (y0 > 0.5f) ? 1.f : 0.f;
  float t1 = thres1[rr];
  float tt1 = 0.05f * (t1 * t1 * t1 + 5.f * t1);
  float y1 = 1.f / (1.f + __expf(-(wn - tt1) * 10.f));
  float m1 = ((y1 > 0.5f) ? 1.f : 0.f) * m0;
  em0[e] = m0 * edge_weight;
  em1[e] = m1 * edge_weight;
}

__global__ __launch_bounds__(256) void deg_kernel(const int* __restrict__ col,
                                                  int* __restrict__ degI, int E) {
  int e = blockIdx.x * 256 + threadIdx.x;
  if (e < E) atomicAdd(&degI[col[e]], 1);
}

__global__ __launch_bounds__(256) void invdeg_kernel(const int* __restrict__ degI,
                                                     float* __restrict__ invdeg, int N) {
  int n = blockIdx.x * 256 + threadIdx.x;
  if (n < N) { int d = degI[n]; invdeg[n] = (d > 0) ? (1.0f / (float)d) : 0.0f; }
}

// ---------------------------------------------------------------------------
// Scatter aggregation: out[col[e], :] += x[row[e], :] * em[e]  (4 dims/thread)
// ---------------------------------------------------------------------------
__global__ __launch_bounds__(256) void agg_kernel(
    const float* __restrict__ x, const int* __restrict__ row, const int* __restrict__ col,
    const float* __restrict__ em, float* __restrict__ out, int E, int D) {
  int idx = blockIdx.x * 256 + threadIdx.x;
  int per = D >> 2;
  int e = idx / per;
  if (e >= E) return;
  int d = (idx - e * per) * 4;
  float w = em[e];
  if (w == 0.f) return;
  const float* src = x + (size_t)row[e] * D + d;
  float* dst = out + (size_t)col[e] * D + d;
  atomicAdd(dst + 0, src[0] * w);
  atomicAdd(dst + 1, src[1] * w);
  atomicAdd(dst + 2, src[2] * w);
  atomicAdd(dst + 3, src[3] * w);
}

// ---------------------------------------------------------------------------
// GIN layer 0: h0 = relu( ((1+eps0)h + neigh*invdeg) @ W0 + b0 )  [N,128]->[N,256]
// 128 nodes/block (8 waves x 16-node tiles); B blocks double-buffered.
// ---------------------------------------------------------------------------
__global__ __launch_bounds__(256) void gin0_kernel(
    const float* __restrict__ h, const float* __restrict__ neigh,
    const float* __restrict__ invdeg, const float* __restrict__ eps0p,
    const _Float16* __restrict__ W0pk, const float* __restrict__ b0,
    float* __restrict__ h0, int N) {
  __shared__ _Float16 xLds[128 * IN_DIM];   // 32 KB
  const int t = threadIdx.x;
  const int n0 = blockIdx.x * 128;
  const float s = 1.0f + eps0p[0];
  {
    int nL = t >> 1, part = t & 1;
    int node = n0 + nL; if (node >= N) node = N - 1;
    float id = invdeg[node];
    const float4* hs = (const float4*)(h + (size_t)node * IN_DIM + part * 64);
    const float4* ns = (const float4*)(neigh + (size_t)node * IN_DIM + part * 64);
    _Float16* dst = &xLds[nL * IN_DIM + part * 64];
#pragma unroll
    for (int j = 0; j < 16; ++j) {
      float4 a = hs[j]; float4 b = ns[j];
      dst[4 * j + 0] = (_Float16)(s * a.x + b.x * id);
      dst[4 * j + 1] = (_Float16)(s * a.y + b.y * id);
      dst[4 * j + 2] = (_Float16)(s * a.z + b.z * id);
      dst[4 * j + 3] = (_Float16)(s * a.w + b.w * id);
    }
  }
  __syncthreads();

  const int lane = t & 31, wave = t >> 5;
  const int g = lane >> 4, mr = lane & 15;
  const _Float16* aRow = &xLds[(wave * 16 + mr) * IN_DIM];
  V16 afrag[4];
#pragma unroll
  for (int kc = 0; kc < 4; ++kc) {
    afrag[kc].h[0] = *(const v8h*)(aRow + kc * 32 + 8 * g);
    afrag[kc].h[1] = *(const v8h*)(aRow + kc * 32 + 16 + 8 * g);
  }

  const _Float16* bptr = W0pk + (size_t)lane * 16;
  auto loadBlock = [&](V16* bbuf, int ncb) {
    const _Float16* nb = bptr + (size_t)ncb * 2048;   // 4 frags * 512 halves
#pragma unroll
    for (int kc = 0; kc < 4; ++kc) {
      bbuf[kc].h[0] = *(const v8h*)(nb + kc * 512);
      bbuf[kc].h[1] = *(const v8h*)(nb + kc * 512 + 8);
    }
  };
  auto computeBlock = [&](const V16* bbuf, int nc) {
    v8f c = {};
#pragma unroll
    for (int kc = 0; kc < 4; ++kc) {
      c = __builtin_amdgcn_wmma_f32_16x16x32_f16(false, afrag[kc].v, false, bbuf[kc].v,
                                                 (short)0, c, false, false);
    }
    float bias = b0[nc * 16 + mr];
#pragma unroll
    for (int r = 0; r < 8; ++r) {
      int node = n0 + wave * 16 + r + 8 * g;
      if (node < N) {
        float v = c[r] + bias;
        h0[(size_t)node * HID + nc * 16 + mr] = (v > 0.f) ? v : 0.f;
      }
    }
  };

  V16 bbA[4], bbB[4];
  loadBlock(bbA, 0);
  for (int nc = 0; nc < HID / 16; nc += 2) {
    loadBlock(bbB, nc + 1);
    computeBlock(bbA, nc);
    if (nc + 2 < HID / 16) loadBlock(bbA, nc + 2);
    computeBlock(bbB, nc + 1);
  }
}

// ---------------------------------------------------------------------------
// GIN layer 1 + prediction head, fused:
// out = ( relu( ((1+eps1)h0 + neigh1*invdeg) @ W1 + b1 ) + h0 @ Wpred ) / 2
// 64 nodes/block (4 waves); W1/Wpred packed with N padded 40 -> 48.
// ---------------------------------------------------------------------------
__global__ __launch_bounds__(128) void gin1_kernel(
    const float* __restrict__ h0, const float* __restrict__ neigh,
    const float* __restrict__ invdeg, const float* __restrict__ eps1p,
    const _Float16* __restrict__ W1pk, const float* __restrict__ b1,
    const _Float16* __restrict__ Wppk, float* __restrict__ out, int N) {
  __shared__ _Float16 x1Lds[64 * HID];   // 32 KB
  __shared__ _Float16 h0Lds[64 * HID];   // 32 KB
  const int t = threadIdx.x;
  const int n0 = blockIdx.x * 64;
  const float s = 1.0f + eps1p[0];
  {
    int nL = t >> 1, part = t & 1;
    int node = n0 + nL; if (node >= N) node = N - 1;
    float id = invdeg[node];
    const float4* hs = (const float4*)(h0 + (size_t)node * HID + part * 128);
    const float4* ns = (const float4*)(neigh + (size_t)node * HID + part * 128);
    _Float16* dx = &x1Lds[nL * HID + part * 128];
    _Float16* dh = &h0Lds[nL * HID + part * 128];
#pragma unroll
    for (int j = 0; j < 32; ++j) {
      float4 a = hs[j]; float4 b = ns[j];
      dh[4 * j + 0] = (_Float16)a.x;  dx[4 * j + 0] = (_Float16)(s * a.x + b.x * id);
      dh[4 * j + 1] = (_Float16)a.y;  dx[4 * j + 1] = (_Float16)(s * a.y + b.y * id);
      dh[4 * j + 2] = (_Float16)a.z;  dx[4 * j + 2] = (_Float16)(s * a.z + b.z * id);
      dh[4 * j + 3] = (_Float16)a.w;  dx[4 * j + 3] = (_Float16)(s * a.w + b.w * id);
    }
  }
  __syncthreads();

  const int lane = t & 31, wave = t >> 5;
  const int g = lane >> 4, mr = lane & 15;
  const _Float16* xRow = &x1Lds[(wave * 16 + mr) * HID];
  const _Float16* hRow = &h0Lds[(wave * 16 + mr) * HID];
  V16 a1[8], a2[8];
#pragma unroll
  for (int kc = 0; kc < 8; ++kc) {
    a1[kc].h[0] = *(const v8h*)(xRow + kc * 32 + 8 * g);
    a1[kc].h[1] = *(const v8h*)(xRow + kc * 32 + 16 + 8 * g);
    a2[kc].h[0] = *(const v8h*)(hRow + kc * 32 + 8 * g);
    a2[kc].h[1] = *(const v8h*)(hRow + kc * 32 + 16 + 8 * g);
  }
  for (int nc = 0; nc < NPAD / 16; ++nc) {
    v8f c1 = {}, c2 = {};
#pragma unroll
    for (int kc = 0; kc < 8; ++kc) {
      size_t boff = (((size_t)(nc * 8 + kc)) * 32 + lane) * 16;
      V16 b1v, b2v;
      b1v.h[0] = *(const v8h*)(W1pk + boff);
      b1v.h[1] = *(const v8h*)(W1pk + boff + 8);
      b2v.h[0] = *(const v8h*)(Wppk + boff);
      b2v.h[1] = *(const v8h*)(Wppk + boff + 8);
      c1 = __builtin_amdgcn_wmma_f32_16x16x32_f16(false, a1[kc].v, false, b1v.v,
                                                  (short)0, c1, false, false);
      c2 = __builtin_amdgcn_wmma_f32_16x16x32_f16(false, a2[kc].v, false, b2v.v,
                                                  (short)0, c2, false, false);
    }
    int ncol = nc * 16 + mr;
    float bias = (ncol < NCLS) ? b1[ncol] : 0.f;
#pragma unroll
    for (int r = 0; r < 8; ++r) {
      int node = n0 + wave * 16 + r + 8 * g;
      if (node < N && ncol < NCLS) {
        float v = c1[r] + bias;
        v = (v > 0.f) ? v : 0.f;
        out[(size_t)node * NCLS + ncol] = 0.5f * (v + c2[r]);
      }
    }
  }
}

// ---------------------------------------------------------------------------
extern "C" void kernel_launch(void* const* d_in, const int* in_sizes, int n_in,
                              void* d_out, int out_size, void* d_ws, size_t ws_size,
                              hipStream_t stream) {
  const float* h      = (const float*)d_in[0];
  const int*   row    = (const int*)d_in[3];
  const int*   col    = (const int*)d_in[4];
  const float* We1    = (const float*)d_in[5];
  const float* be1    = (const float*)d_in[6];
  const float* We2    = (const float*)d_in[7];
  const float* be2    = (const float*)d_in[8];
  const float* eps0   = (const float*)d_in[9];
  const float* W0     = (const float*)d_in[10];
  const float* b0     = (const float*)d_in[11];
  const float* eps1   = (const float*)d_in[12];
  const float* W1     = (const float*)d_in[13];
  const float* b1     = (const float*)d_in[14];
  const float* thres0 = (const float*)d_in[15];
  const float* thres1 = (const float*)d_in[16];
  const float* Wpred  = (const float*)d_in[17];
  const int E = in_sizes[3];
  const int N = in_sizes[15];
  float* out = (float*)d_out;

  // workspace carve-out
  char* ws = (char*)d_ws;
  size_t off = 0;
  auto alloc = [&](size_t bytes) -> void* {
    void* p = ws + off;
    off = (off + bytes + 255) & ~(size_t)255;
    return p;
  };
  float*    stats  = (float*)alloc(64);
  _Float16* We1p   = (_Float16*)alloc((size_t)2 * IN_DIM * 2 * EL_HID);
  _Float16* W0p    = (_Float16*)alloc((size_t)IN_DIM * HID * 2);
  _Float16* W1p    = (_Float16*)alloc((size_t)HID * NPAD * 2);
  _Float16* Wpp    = (_Float16*)alloc((size_t)HID * NPAD * 2);
  float*    ew     = (float*)alloc((size_t)E * 4);
  float*    em0    = (float*)alloc((size_t)E * 4);
  float*    em1    = (float*)alloc((size_t)E * 4);
  int*      degI   = (int*)alloc((size_t)N * 4);
  float*    invdeg = (float*)alloc((size_t)N * 4);
  float*    neigh0 = (float*)alloc((size_t)N * IN_DIM * 4);
  float*    h0buf  = (float*)alloc((size_t)N * HID * 4);
  float*    neigh1 = (float*)alloc((size_t)N * HID * 4);

  // 1) pack weights to WMMA-B f16 layout
  {
    int tot = 2 * IN_DIM * EL_HID;
    pack_b_kernel<<<(tot + 255) / 256, 256, 0, stream>>>(We1, We1p, 2 * IN_DIM, EL_HID, EL_HID);
  }
  {
    int tot = IN_DIM * HID;
    pack_b_kernel<<<(tot + 255) / 256, 256, 0, stream>>>(W0, W0p, IN_DIM, HID, HID);
  }
  {
    int tot = HID * NPAD;
    pack_b_kernel<<<(tot + 255) / 256, 256, 0, stream>>>(W1, W1p, HID, NCLS, NPAD);
    pack_b_kernel<<<(tot + 255) / 256, 256, 0, stream>>>(Wpred, Wpp, HID, NCLS, NPAD);
  }

  // 2) zero accumulators
  hipMemsetAsync(degI,   0, (size_t)N * 4, stream);
  hipMemsetAsync(neigh0, 0, (size_t)N * IN_DIM * 4, stream);
  hipMemsetAsync(neigh1, 0, (size_t)N * HID * 4, stream);

  // 3) edge MLP (dominant GEMM, WMMA f16, double-buffered B)
  edge_mlp_kernel<<<(E + 127) / 128, 128, 0, stream>>>(h, row, col, We1p, be1, We2, be2, ew, E);

  // 4) edge-weight stats + masks
  stats_kernel<<<1, 1024, 0, stream>>>(ew, stats, E);
  mask_kernel<<<(E + 255) / 256, 256, 0, stream>>>(ew, stats, row, thres0, thres1, em0, em1, E);

  // 5) degrees
  deg_kernel<<<(E + 255) / 256, 256, 0, stream>>>(col, degI, E);
  invdeg_kernel<<<(N + 255) / 256, 256, 0, stream>>>(degI, invdeg, N);

  // 6) layer 0: aggregate + GEMM
  {
    long long tot = (long long)E * (IN_DIM / 4);
    agg_kernel<<<(unsigned)((tot + 255) / 256), 256, 0, stream>>>(h, row, col, em0, neigh0, E, IN_DIM);
  }
  gin0_kernel<<<(N + 127) / 128, 256, 0, stream>>>(h, neigh0, invdeg, eps0, W0p, b0, h0buf, N);

  // 7) layer 1: aggregate + fused GEMM + prediction head
  {
    long long tot = (long long)E * (HID / 4);
    agg_kernel<<<(unsigned)((tot + 255) / 256), 256, 0, stream>>>(h0buf, row, col, em1, neigh1, E, HID);
  }
  gin1_kernel<<<(N + 63) / 64, 128, 0, stream>>>(h0buf, neigh1, invdeg, eps1, W1p, b1, Wpp, out, N);
}
